// DeformableDecoderLayer_9174050144403
// MI455X (gfx1250) — matrix operand order repaired and checked by reading, not compiled, not run on hardware
//
#include <hip/hip_runtime.h>
#include <stdint.h>

#define B_   16
#define Q_   300
#define D_   256
#define H_   8
#define P_   4
#define F_   2048
#define HG_  100
#define WG_  100
#define HW_  (HG_ * WG_)
#define DH_  32
#define EPS_ 1e-5f

typedef __attribute__((ext_vector_type(16))) _Float16 v16h;
typedef __attribute__((ext_vector_type(8)))  _Float16 v8h;
typedef __attribute__((ext_vector_type(8)))  float    v8f;

// 128-bit payload type for the async global->LDS builtins (matches the
// builtin's parameter type: GCC-vector of 4 ints).
typedef int v4i __attribute__((vector_size(16)));
typedef __attribute__((address_space(1))) v4i GV4;   // global int4
typedef __attribute__((address_space(3))) v4i LV4;   // LDS int4

// Use gfx1250 async global->LDS copies when the toolchain exposes them.
#if defined(__HIP_DEVICE_COMPILE__) && __has_builtin(__builtin_amdgcn_global_load_async_to_lds_b128)
#define ASYNC_LDS 1
#endif

__device__ __forceinline__ void wait_async0() {
#if defined(__HIP_DEVICE_COMPILE__)
 #if __has_builtin(__builtin_amdgcn_s_wait_asynccnt)
    __builtin_amdgcn_s_wait_asynccnt(0);
 #else
    asm volatile("s_wait_asynccnt 0x0" ::: "memory");
 #endif
#endif
}

__device__ __forceinline__ GV4* as_gv(const void* p) {
    return (GV4*)(uintptr_t)p;
}
__device__ __forceinline__ LV4* as_lv(const void* p) {
    // generic shared address = {aperture_hi, lds_offset[31:0]} -> low 32 bits
    return (LV4*)(unsigned)(uintptr_t)p;
}

// 16-bit A-matrix 16x32 layout (ISA 7.12.2): lanes 0-15 hold K={0..7,16..23},
// lanes 16-31 hold K={8..15,24..31}; M = lane & 15.
__device__ __forceinline__ int a_kidx(int lane, int j) {
    return ((lane & 16) ? 8 : 0) + ((j < 8) ? j : (j + 8));
}
// B-matrix 32x16 (same K order as A): lanes 0-15 hold K={0..7,16..23},
// lanes 16-31 hold K={8..15,24..31}; N = lane & 15.
__device__ __forceinline__ int b_kidx(int lane, int j) {
    return ((lane & 16) ? 8 : 0) + ((j < 8) ? j : (j + 8));
}

__device__ __forceinline__ v16h mk16(v8h lo, v8h hi) {
    v16h r;
#pragma unroll
    for (int j = 0; j < 8; ++j) { r[j] = lo[j]; r[j + 8] = hi[j]; }
    return r;
}

// ---------------------------------------------------------------------------
// Generic GEMM: C[M,N] = A[M,K] @ W[K,N] + bias (optional ReLU).
// fp32 inputs, f16 WMMA compute, f32 accumulate, OutT output (float/_Float16).
// Block: 256 threads = 8 waves; block tile 64x64, wave tile 16x32, BK=32.
// Requires M%64==0 (async A staging is unguarded), N%64==0, K%32==0 as
// launched below.
// ---------------------------------------------------------------------------
template <bool RELU, typename OutT>
__global__ __launch_bounds__(256) void wmma_gemm(
    const float* __restrict__ A, const float* __restrict__ Bw,
    const float* __restrict__ bias, OutT* __restrict__ C,
    int M, int N, int K)
{
#ifdef ASYNC_LDS
    __shared__ alignas(64) float    As32[64][32];  // [m][k], staged via async DMA
#else
    __shared__ alignas(16) _Float16 As[64][32];    // [m][k]
#endif
    __shared__ alignas(16) _Float16 Bs[64][32];    // [n][k] (transposed)

    const int tid  = threadIdx.x;
    const int lane = tid & 31;
    const int wave = tid >> 5;
    const int wm   = wave >> 1;          // 0..3 -> M sub-tile
    const int wn   = wave & 1;           // 0..1 -> N sub-tile (32 wide)
    const int m_blk = blockIdx.x * 64;
    const int n_blk = blockIdx.y * 64;
    const int r    = lane & 15;

    v8f acc0 = {};
    v8f acc1 = {};

    for (int k0 = 0; k0 < K; k0 += 32) {
#ifdef ASYNC_LDS
        // stage A tile 64x32 f32 with async global->LDS b128 copies:
        // wave w owns rows [8w, 8w+8); each lane moves one 16B chunk per issue.
        {
            int row = wave * 8 + (lane >> 3);
            int c4  = (lane & 7) * 4;
            const float* g0 = &A[(size_t)(m_blk + row) * K + (k0 + c4)];
            const float* g1 = &A[(size_t)(m_blk + row + 4) * K + (k0 + c4)];
            __builtin_amdgcn_global_load_async_to_lds_b128(
                as_gv(g0), as_lv(&As32[row][c4]), 0, 0);
            __builtin_amdgcn_global_load_async_to_lds_b128(
                as_gv(g1), as_lv(&As32[row + 4][c4]), 0, 0);
        }
#else
        // synchronous fallback: stage A tile 64x32 (f32 -> f16)
#pragma unroll
        for (int i = 0; i < 8; ++i) {
            int e  = tid + i * 256;
            int rr = e >> 5, kk = e & 31;
            int gm = m_blk + rr;
            float v = (gm < M) ? A[(size_t)gm * K + (k0 + kk)] : 0.f;
            As[rr][kk] = (_Float16)v;
        }
#endif
        // stage B tile 32x64, transposed into LDS (f32 -> f16)
#pragma unroll
        for (int i = 0; i < 8; ++i) {
            int e  = tid + i * 256;
            int kk = e >> 6, nn = e & 63;
            int gn = n_blk + nn;
            float v = (gn < N) ? Bw[(size_t)(k0 + kk) * N + gn] : 0.f;
            Bs[nn][kk] = (_Float16)v;
        }
#ifdef ASYNC_LDS
        wait_async0();
#endif
        __syncthreads();

        const int akb = (lane & 16) ? 8 : 0;
#ifdef ASYNC_LDS
        v16h a;
        {
            const float* ap = &As32[wm * 16 + r][0];
            v8f flo = *(const v8f*)&ap[akb];         // 32B contiguous
            v8f fhi = *(const v8f*)&ap[akb + 16];
#pragma unroll
            for (int j = 0; j < 8; ++j) {
                a[j]     = (_Float16)flo[j];
                a[j + 8] = (_Float16)fhi[j];
            }
        }
#else
        v16h a = mk16(*(const v8h*)&As[wm * 16 + r][akb],
                      *(const v8h*)&As[wm * 16 + r][akb + 16]);
#endif
        const int bkb = (lane & 16) ? 8 : 0;
        v16h b0 = mk16(*(const v8h*)&Bs[wn * 32 + r][bkb],
                       *(const v8h*)&Bs[wn * 32 + r][bkb + 16]);
        v16h b1 = mk16(*(const v8h*)&Bs[wn * 32 + 16 + r][bkb],
                       *(const v8h*)&Bs[wn * 32 + 16 + r][bkb + 16]);

        acc0 = __builtin_amdgcn_wmma_f32_16x16x32_f16(false, a, false, b0,
                                                      (short)0, acc0, false, false);
        acc1 = __builtin_amdgcn_wmma_f32_16x16x32_f16(false, a, false, b1,
                                                      (short)0, acc1, false, false);
        __syncthreads();
    }

    // C/D layout: lane l, vgpr rr: N = l&15, M = rr + (l<16 ? 0 : 8)
    const int mo = (lane & 16) ? 8 : 0;
#pragma unroll
    for (int rr = 0; rr < 8; ++rr) {
        int m = m_blk + wm * 16 + rr + mo;
        if (m >= M) continue;
        int n0 = n_blk + wn * 32 + r;
        int n1 = n0 + 16;
        float v0 = acc0[rr] + (bias ? bias[n0] : 0.f);
        float v1 = acc1[rr] + (bias ? bias[n1] : 0.f);
        if (RELU) { v0 = fmaxf(v0, 0.f); v1 = fmaxf(v1, 0.f); }
        if (n0 < N) C[(size_t)m * N + n0] = (OutT)v0;
        if (n1 < N) C[(size_t)m * N + n1] = (OutT)v1;
    }
}

// ---------------------------------------------------------------------------
// Multi-head self-attention from packed qkv [B,Q,3D].
// Block: 64 threads = 2 waves; each wave handles 16 query rows of one (b,h).
// Grid: (B*H, ceil(Q/32)).
// ---------------------------------------------------------------------------
__global__ __launch_bounds__(64) void attn_kernel(
    const float* __restrict__ qkv, float* __restrict__ sa)
{
    __shared__ float probs[2][16][320];   // keys padded 300 -> 320

    const int tid  = threadIdx.x;
    const int lane = tid & 31;
    const int wv   = tid >> 5;
    const int b    = blockIdx.x >> 3;
    const int h    = blockIdx.x & 7;
    const int q0   = (blockIdx.y * 2 + wv) * 16;
    const int r    = lane & 15;
    const float scale = 0.17677669529663687f;   // 1/sqrt(32)
    const size_t base = (size_t)b * Q_ * (3 * D_);

    // zero the softmax pad region (keys 300..319)
    for (int i = lane; i < 16 * 20; i += 32) {
        probs[wv][i / 20][300 + (i % 20)] = 0.f;
    }

    // q fragment: 16 rows x K=32 (one WMMA K-step)
    v16h qa;
    {
        int qrow = q0 + r; if (qrow > Q_ - 1) qrow = Q_ - 1;
        const float* qp = qkv + base + (size_t)qrow * (3 * D_) + h * DH_;
#pragma unroll
        for (int j = 0; j < 16; ++j) qa[j] = (_Float16)qp[a_kidx(lane, j)];
    }

    // scores: 19 key tiles of 16
    for (int kt = 0; kt < 19; ++kt) {
        int key  = kt * 16 + r;
        int keyc = key < Q_ ? key : Q_ - 1;
        const float* kp = qkv + base + (size_t)keyc * (3 * D_) + D_ + h * DH_;
        v16h kb;
#pragma unroll
        for (int j = 0; j < 16; ++j) kb[j] = (_Float16)kp[b_kidx(lane, j)];
        v8f s = {};
        s = __builtin_amdgcn_wmma_f32_16x16x32_f16(false, qa, false, kb,
                                                   (short)0, s, false, false);
        int mo = (lane & 16) ? 8 : 0;
#pragma unroll
        for (int rr = 0; rr < 8; ++rr) {
            if (key < Q_) probs[wv][rr + mo][key] = s[rr] * scale;
        }
    }
    __syncthreads();

    // row-wise softmax over 300 keys (lanes 0..15 each own a row)
    if (lane < 16) {
        float mx = -1e30f;
        for (int j = 0; j < Q_; ++j) mx = fmaxf(mx, probs[wv][lane][j]);
        float sum = 0.f;
        for (int j = 0; j < Q_; ++j) {
            float e = __expf(probs[wv][lane][j] - mx);
            probs[wv][lane][j] = e;
            sum += e;
        }
        float inv = 1.f / sum;
        for (int j = 0; j < Q_; ++j) probs[wv][lane][j] *= inv;
    }
    __syncthreads();

    // attn @ v : K padded to 320 (pad probs are exactly 0)
    v8f o0 = {}, o1 = {};
    for (int kc = 0; kc < 10; ++kc) {
        v16h pa;
#pragma unroll
        for (int j = 0; j < 16; ++j)
            pa[j] = (_Float16)probs[wv][r][kc * 32 + a_kidx(lane, j)];
        v16h vb0, vb1;
#pragma unroll
        for (int j = 0; j < 16; ++j) {
            int key  = kc * 32 + b_kidx(lane, j);
            int keyc = key < Q_ ? key : Q_ - 1;
            const float* vp = qkv + base + (size_t)keyc * (3 * D_) + 2 * D_ + h * DH_;
            vb0[j] = (_Float16)vp[r];
            vb1[j] = (_Float16)vp[16 + r];
        }
        o0 = __builtin_amdgcn_wmma_f32_16x16x32_f16(false, pa, false, vb0,
                                                    (short)0, o0, false, false);
        o1 = __builtin_amdgcn_wmma_f32_16x16x32_f16(false, pa, false, vb1,
                                                    (short)0, o1, false, false);
    }

    const int mo = (lane & 16) ? 8 : 0;
#pragma unroll
    for (int rr = 0; rr < 8; ++rr) {
        int qrow = q0 + rr + mo;
        if (qrow < Q_) {
            size_t off = ((size_t)b * Q_ + qrow) * D_ + h * DH_;
            sa[off + r]      = o0[rr];
            sa[off + 16 + r] = o1[rr];
        }
    }
}

// ---------------------------------------------------------------------------
// out = LayerNorm(xa + xb) * g + beta ; one 256-thread block per row (D=256)
// ---------------------------------------------------------------------------
__global__ __launch_bounds__(256) void ln_kernel(
    const float* __restrict__ xa, const float* __restrict__ xb,
    const float* __restrict__ g, const float* __restrict__ be,
    float* __restrict__ out)
{
    __shared__ float red[256];
    const int row = blockIdx.x, t = threadIdx.x;
    float v = xa[(size_t)row * D_ + t] + xb[(size_t)row * D_ + t];
    red[t] = v;
    __syncthreads();
    for (int s = 128; s > 0; s >>= 1) { if (t < s) red[t] += red[t + s]; __syncthreads(); }
    float mean = red[0] / D_;
    __syncthreads();
    float d = v - mean;
    red[t] = d * d;
    __syncthreads();
    for (int s = 128; s > 0; s >>= 1) { if (t < s) red[t] += red[t + s]; __syncthreads(); }
    float var = red[0] / D_;
    out[(size_t)row * D_ + t] = d * rsqrtf(var + EPS_) * g[t] + be[t];
}

// ---------------------------------------------------------------------------
// ref / offsets / attention-weight projections, fused sigmoid + clip + softmax.
// One 128-thread block per (b,q) row. Emits loc[B,Q,H,P,2] and wts[B,Q,H,P].
// ---------------------------------------------------------------------------
__global__ __launch_bounds__(128) void refoff_kernel(
    const float* __restrict__ x,
    const float* __restrict__ ref_w, const float* __restrict__ ref_b,
    const float* __restrict__ off_w, const float* __restrict__ off_b,
    const float* __restrict__ attw_w, const float* __restrict__ attw_b,
    float* __restrict__ loc, float* __restrict__ wts)
{
    __shared__ float xs[256];
    __shared__ float refs[2];
    __shared__ float offs[64];
    __shared__ float aws[32];
    const int row = blockIdx.x, t = threadIdx.x;
    xs[t]       = x[(size_t)row * D_ + t];
    xs[t + 128] = x[(size_t)row * D_ + t + 128];
    __syncthreads();

    if (t < 2) {
        float s = ref_b[t];
        for (int k = 0; k < D_; ++k) s += xs[k] * ref_w[k * 2 + t];
        refs[t] = 1.f / (1.f + __expf(-s));
    } else if (t < 66) {
        int i = t - 2;
        float s = off_b[i];
        for (int k = 0; k < D_; ++k) s += xs[k] * off_w[k * 64 + i];
        offs[i] = s;
    } else if (t < 98) {
        int i = t - 66;
        float s = attw_b[i];
        for (int k = 0; k < D_; ++k) s += xs[k] * attw_w[k * 32 + i];
        aws[i] = s;
    }
    __syncthreads();

    if (t < 8) {  // softmax over P=4 per head
        float a0 = aws[t * 4 + 0], a1 = aws[t * 4 + 1];
        float a2 = aws[t * 4 + 2], a3 = aws[t * 4 + 3];
        float mx = fmaxf(fmaxf(a0, a1), fmaxf(a2, a3));
        float e0 = __expf(a0 - mx), e1 = __expf(a1 - mx);
        float e2 = __expf(a2 - mx), e3 = __expf(a3 - mx);
        float inv = 1.f / (e0 + e1 + e2 + e3);
        size_t o = ((size_t)row * 8 + t) * 4;
        wts[o + 0] = e0 * inv; wts[o + 1] = e1 * inv;
        wts[o + 2] = e2 * inv; wts[o + 3] = e3 * inv;
    }
    if (t < 64) {  // loc = clip(ref + offset)
        int h = t >> 3, p = (t >> 1) & 3, c = t & 1;
        float l = refs[c] + offs[(h * 4 + p) * 2 + c];
        l = fminf(fmaxf(l, 0.f), 1.f);
        loc[(((size_t)row * 8 + h) * 4 + p) * 2 + c] = l;
    }
}

// ---------------------------------------------------------------------------
// Deformable bilinear sampling + weighted sum over P points (f16 values).
// One 256-thread block per (b,q): thread = (h, dh).
// ---------------------------------------------------------------------------
__global__ __launch_bounds__(256) void sample_kernel(
    const _Float16* __restrict__ vals, const float* __restrict__ loc,
    const float* __restrict__ wts, float* __restrict__ ca)
{
    const int row = blockIdx.x;          // b*Q + q
    const int b   = row / Q_;
    const int t   = threadIdx.x;
    const int h   = t >> 5, dh = t & 31;

    float acc = 0.f;
#pragma unroll
    for (int p = 0; p < P_; ++p) {
        size_t li = ((size_t)row * 8 + h) * 4 + p;
        float lx = loc[li * 2 + 0], ly = loc[li * 2 + 1];
        float w  = wts[li];
        float sx = lx * (WG_ - 1), sy = ly * (HG_ - 1);
        float x0f = floorf(sx), y0f = floorf(sy);
        int x0 = (int)x0f, y0 = (int)y0f;
        x0 = x0 < 0 ? 0 : (x0 > WG_ - 1 ? WG_ - 1 : x0);
        y0 = y0 < 0 ? 0 : (y0 > HG_ - 1 ? HG_ - 1 : y0);
        int x1 = x0 + 1 > WG_ - 1 ? WG_ - 1 : x0 + 1;
        int y1 = y0 + 1 > HG_ - 1 ? HG_ - 1 : y0 + 1;
        float wx1 = sx - x0f, wx0 = 1.f - wx1;
        float wy1 = sy - y0f, wy0 = 1.f - wy1;
        const _Float16* vb = vals + (size_t)b * HW_ * D_ + h * DH_ + dh;
        float v00 = (float)vb[(size_t)(y0 * WG_ + x0) * D_];
        float v01 = (float)vb[(size_t)(y1 * WG_ + x0) * D_];
        float v10 = (float)vb[(size_t)(y0 * WG_ + x1) * D_];
        float v11 = (float)vb[(size_t)(y1 * WG_ + x1) * D_];
        acc += w * (v00 * wx0 * wy0 + v01 * wx0 * wy1 +
                    v10 * wx1 * wy0 + v11 * wx1 * wy1);
    }
    ca[(size_t)row * D_ + h * DH_ + dh] = acc;
}

// ---------------------------------------------------------------------------
extern "C" void kernel_launch(void* const* d_in, const int* in_sizes, int n_in,
                              void* d_out, int out_size, void* d_ws, size_t ws_size,
                              hipStream_t stream)
{
    (void)in_sizes; (void)n_in; (void)out_size; (void)ws_size;

    const float* tgt        = (const float*)d_in[0];
    const float* memory_    = (const float*)d_in[1];
    const float* in_proj_w  = (const float*)d_in[2];
    const float* in_proj_b  = (const float*)d_in[3];
    const float* out_proj_w = (const float*)d_in[4];
    const float* out_proj_b = (const float*)d_in[5];
    const float* norm1_g    = (const float*)d_in[6];
    const float* norm1_b    = (const float*)d_in[7];
    const float* ref_w      = (const float*)d_in[8];
    const float* ref_b      = (const float*)d_in[9];
    const float* off_w      = (const float*)d_in[10];
    const float* off_b      = (const float*)d_in[11];
    const float* attw_w     = (const float*)d_in[12];
    const float* attw_b     = (const float*)d_in[13];
    const float* vproj_w    = (const float*)d_in[14];
    const float* vproj_b    = (const float*)d_in[15];
    const float* oproj_w    = (const float*)d_in[16];
    const float* oproj_b    = (const float*)d_in[17];
    const float* norm2_g    = (const float*)d_in[18];
    const float* norm2_b    = (const float*)d_in[19];
    const float* lin1_w     = (const float*)d_in[20];
    const float* lin1_b     = (const float*)d_in[21];
    const float* lin2_w     = (const float*)d_in[22];
    const float* lin2_b     = (const float*)d_in[23];
    const float* norm3_g    = (const float*)d_in[24];
    const float* norm3_b    = (const float*)d_in[25];

    float* ws = (float*)d_ws;
    size_t off = 0;
    auto alloc = [&](size_t n) { float* p = ws + off; off += n; return p; };
    const int BQ = B_ * Q_;                              // 4800
    float* qkv  = alloc((size_t)BQ * 768);               // 14.7 MB
    float* sa   = alloc((size_t)BQ * D_);
    float* tmp  = alloc((size_t)BQ * D_);
    float* x1   = alloc((size_t)BQ * D_);
    float* loc  = alloc((size_t)BQ * H_ * P_ * 2);
    float* wts  = alloc((size_t)BQ * H_ * P_);
    _Float16* vals = (_Float16*)alloc(((size_t)B_ * HW_ * D_) / 2);  // 81.9 MB f16
    float* ca   = alloc((size_t)BQ * D_);
    float* x2   = alloc((size_t)BQ * D_);
    float* ff   = alloc((size_t)BQ * F_);                // 39.3 MB
    float* out  = (float*)d_out;

    // 1. QKV projection [4800,256] @ [256,768]
    wmma_gemm<false, float><<<dim3(75, 12), 256, 0, stream>>>(
        tgt, in_proj_w, in_proj_b, qkv, BQ, 3 * D_, D_);
    // 2. multi-head self-attention
    attn_kernel<<<dim3(B_ * H_, 10), 64, 0, stream>>>(qkv, sa);
    // 3. out projection
    wmma_gemm<false, float><<<dim3(75, 4), 256, 0, stream>>>(
        sa, out_proj_w, out_proj_b, tmp, BQ, D_, D_);
    // 4. x = LN(tgt + sa)
    ln_kernel<<<BQ, 256, 0, stream>>>(tgt, tmp, norm1_g, norm1_b, x1);
    // 5. ref points / offsets / attention weights
    refoff_kernel<<<BQ, 128, 0, stream>>>(x1, ref_w, ref_b, off_w, off_b,
                                          attw_w, attw_b, loc, wts);
    // 6. value projection of memory [160000,256] @ [256,256] -> f16 values
    wmma_gemm<false, _Float16><<<dim3(2500, 4), 256, 0, stream>>>(
        memory_, vproj_w, vproj_b, vals, B_ * HW_, D_, D_);
    // 7. deformable bilinear sampling
    sample_kernel<<<BQ, 256, 0, stream>>>(vals, loc, wts, ca);
    // 8. output projection of cross-attention
    wmma_gemm<false, float><<<dim3(75, 4), 256, 0, stream>>>(
        ca, oproj_w, oproj_b, tmp, BQ, D_, D_);
    // 9. x = LN(x + ca)
    ln_kernel<<<BQ, 256, 0, stream>>>(x1, tmp, norm2_g, norm2_b, x2);
    // 10. FFN lin1 + ReLU [4800,256] @ [256,2048]
    wmma_gemm<true, float><<<dim3(75, 32), 256, 0, stream>>>(
        x2, lin1_w, lin1_b, ff, BQ, F_, D_);
    // 11. FFN lin2 [4800,2048] @ [2048,256]
    wmma_gemm<false, float><<<dim3(75, 4), 256, 0, stream>>>(
        ff, lin2_w, lin2_b, tmp, BQ, D_, F_);
    // 12. out = LN(x + ffn)
    ln_kernel<<<BQ, 256, 0, stream>>>(x2, tmp, norm3_g, norm3_b, out);
}
